// ShowAttendLSTM_15255723835405
// MI455X (gfx1250) — matrix operand order repaired and checked by reading, not compiled, not run on hardware
//
#include <hip/hip_runtime.h>
#include <hip/hip_bf16.h>
#include <stddef.h>
#include <stdint.h>

// Show-Attend-Tell LSTM decoder, MI455X (gfx1250).
// bf16 WMMA GEMMs with async global->LDS weight staging (double buffered),
// f32 accumulation; VALU kernels for attention / LSTM cell / log-softmax.

#define V_ 10000
#define E_ 512
#define D_ 1024
#define L_ 196
#define B_ 64
#define T_ 20
#define KXC 2560   // E + 2D  (xc = [e | ctx | h])
#define KIH 1536   // E + D

typedef __attribute__((ext_vector_type(16))) __bf16 v16bf;
typedef __attribute__((ext_vector_type(8)))  float  v8f;

__device__ __forceinline__ unsigned short f2bf(float f) {
    union { float f; unsigned int u; } x; x.f = f;
    unsigned int u = x.u;
    u += 0x7fffu + ((u >> 16) & 1u);          // round-to-nearest-even
    return (unsigned short)(u >> 16);
}
__device__ __forceinline__ float bf2f(unsigned short h) {
    union { unsigned int u; float f; } x; x.u = ((unsigned int)h) << 16;
    return x.f;
}

// Async copy 16B global -> LDS (per-lane), tracked by ASYNCcnt.
// dsaddr = low 32 bits of the generic LDS address (hardware truncates the same way).
__device__ __forceinline__ void async_copy_b128(const void* gaddr, void* lds_ptr) {
    unsigned loff = (unsigned)(uintptr_t)lds_ptr;
    unsigned long long ga = (unsigned long long)(uintptr_t)gaddr;
    asm volatile("global_load_async_to_lds_b128 %0, %1, off"
                 :: "v"(loff), "v"(ga) : "memory");
}
__device__ __forceinline__ void wait_async0() {
    asm volatile("s_wait_asynccnt 0x0" ::: "memory");
}

// ---------------------------------------------------------------- prep kernels
__global__ void k_f32_to_bf16(unsigned short* __restrict__ dst,
                              const float* __restrict__ src, long n) {
    long i = (long)blockIdx.x * blockDim.x + threadIdx.x;
    if (i < n) dst[i] = f2bf(src[i]);
}

// Wg[N=4096, K=2560] = [W_ih | W_hh] concatenated along K
__global__ void k_build_wg(unsigned short* __restrict__ Wg,
                           const float* __restrict__ Wih,
                           const float* __restrict__ Whh, long total) {
    long i = (long)blockIdx.x * blockDim.x + threadIdx.x;
    if (i >= total) return;
    int n = (int)(i / KXC), k = (int)(i % KXC);
    float v = (k < KIH) ? Wih[(long)n * KIH + k] : Whh[(long)n * D_ + (k - KIH)];
    Wg[i] = f2bf(v);
}

__global__ void k_bias_sum(float* __restrict__ bg, const float* __restrict__ bih,
                           const float* __restrict__ bhh, int n) {
    int i = blockIdx.x * blockDim.x + threadIdx.x;
    if (i < n) bg[i] = bih[i] + bhh[i];
}

// c_ws = c0 ; xc[:, 1536:2560] = bf16(h0)
__global__ void k_init_state(float* __restrict__ c_ws, unsigned short* __restrict__ xc,
                             const float* __restrict__ h0, const float* __restrict__ c0) {
    int i = blockIdx.x * blockDim.x + threadIdx.x;   // 64*1024
    if (i >= B_ * D_) return;
    int b = i >> 10, d = i & (D_ - 1);
    c_ws[i] = c0[i];
    xc[(long)b * KXC + KIH + d] = f2bf(h0[i]);
}

// ------------------------------------------------------------- per-step kernels
// xc[:, 0:512] = emb_bf16[tok]
__global__ void k_embed(unsigned short* __restrict__ xc,
                        const unsigned short* __restrict__ emb_bf,
                        const int* __restrict__ tokens, int t) {
    int i = blockIdx.x * blockDim.x + threadIdx.x;   // 64*512
    if (i >= B_ * E_) return;
    int b = i >> 9, j = i & (E_ - 1);
    int tok = tokens[b * T_ + t];
    xc[(long)b * KXC + j] = emb_bf[(long)tok * E_ + j];
}

// GEMM: C[64, N] = A[64, K](bf16, row stride lda) x W[N, K](bf16)^T + bias
// block = 128 thr (4 waves); wave w -> M-tile rows [16w,16w+16); block -> 4 N-tiles.
// Weight slabs (64 N-rows x 64 K) are async-DMA'd into LDS once per block,
// double buffered: slab (i+1) DMA overlaps slab (i) WMMAs. B fragments for one
// slab are preloaded into distinct registers so the 16 ds_load_b128 burst is
// followed by a single dscnt wait and 8 back-to-back WMMAs.
#define KSLAB 64
#define BROW  72   // 64 + 8 bf16 pad => 144B row stride, conflict-free ds_load_b128

template <bool STORE_BF16>
__global__ __launch_bounds__(128)
void k_gemm_wmma(const unsigned short* __restrict__ A, int lda,
                 const unsigned short* __restrict__ W,
                 const float* __restrict__ bias,
                 void* __restrict__ Cout, int N, int K) {
    __shared__ __align__(16) unsigned short bslab[2][64][BROW];

    const int tid  = threadIdx.x;
    const int lane = tid & 31;
    const int wave = tid >> 5;
    const int m0   = wave * 16;
    const int half = lane >> 4;       // 0: lanes 0-15, 1: lanes 16-31
    const int r    = lane & 15;
    const int n0   = blockIdx.x * 64;
    const int nLast = N - 1;

    union AV { v16bf v; uint4 q[2]; };
    v8f acc[4];
#pragma unroll
    for (int j = 0; j < 4; ++j) acc[j] = (v8f)0.0f;

    const unsigned short* arow = A + (size_t)(m0 + r) * lda;

    // stage a 64x64 weight slab into bslab[buf]: 512 16B chunks, 4 per thread
    auto stage = [&](int buf, int kbase) {
#pragma unroll
        for (int qq = 0; qq < 4; ++qq) {
            int c   = qq * 128 + tid;   // 0..511
            int row = c >> 3;           // 0..63  (N column within block tile)
            int kc  = c & 7;            // 16B chunk within the 128B K-row
            int n   = n0 + row;
            n = (n < nLast) ? n : nLast;
            async_copy_b128(W + (size_t)n * K + kbase + kc * 8,
                            &bslab[buf][row][kc * 8]);
        }
    };

    const int nIter = K / KSLAB;        // K is 1024 or 2560 -> divisible by 64
    const int kMaxSlab = K - KSLAB;
    stage(0, 0);
    for (int it = 0; it < nIter; ++it) {
        const int kbase = it * KSLAB;
        wait_async0();                  // my async chunks for slab `it` landed
        __syncthreads();                // everyone's chunks landed; prev reads done

        // branchless prefetch of next slab (last iter redundantly re-stages the
        // final slab into the spare buffer; drained by wait-idle at endpgm)
        int knext = kbase + KSLAB;
        knext = (knext < kMaxSlab) ? knext : kMaxSlab;
        stage((it + 1) & 1, knext);

        // A tiles for both 32-wide K substeps (per-wave unique, direct global)
        uint4 a00 = *(const uint4*)(arow + kbase + half * 8);
        uint4 a01 = *(const uint4*)(arow + kbase + 16 + half * 8);
        uint4 a10 = *(const uint4*)(arow + kbase + 32 + half * 8);
        uint4 a11 = *(const uint4*)(arow + kbase + 48 + half * 8);

        // preload all 8 B fragments (distinct registers -> one ds wait, WMMA burst)
        const unsigned short (*slab)[BROW] = bslab[it & 1];
        AV bfr[8];
#pragma unroll
        for (int ks = 0; ks < 2; ++ks) {
#pragma unroll
            for (int j = 0; j < 4; ++j) {
                // B 32x16 fragment: lane<16 K ks*32+{0..7,16..23} of col n; lane>=16 K +{8..15,24..31}
                const unsigned short* bp = &slab[j * 16 + r][ks * 32 + half * 8];
                bfr[ks * 4 + j].q[0] = *(const uint4*)(bp);
                bfr[ks * 4 + j].q[1] = *(const uint4*)(bp + 16);
            }
        }
#pragma unroll
        for (int ks = 0; ks < 2; ++ks) {
            AV a;
            a.q[0] = ks ? a10 : a00;
            a.q[1] = ks ? a11 : a01;
#pragma unroll
            for (int j = 0; j < 4; ++j) {
                acc[j] = __builtin_amdgcn_wmma_f32_16x16x32_bf16(
                    false, a.v, false, bfr[ks * 4 + j].v, (short)0, acc[j],
                    false, false);
            }
        }
        __syncthreads();                // finish reading slab before it is re-DMA'd
    }

    // C/D layout: VGPR e: lanes 0-15 -> (M=m0+e, N=n0+16j+lane); lanes16-31 -> M=m0+8+e.
#pragma unroll
    for (int j = 0; j < 4; ++j) {
        int n = n0 + j * 16 + r;
        if (n >= N) continue;
        float bv = bias ? bias[n] : 0.0f;
#pragma unroll
        for (int e = 0; e < 8; ++e) {
            int m = m0 + half * 8 + e;
            float v = acc[j][e] + bv;
            if (STORE_BF16)
                ((unsigned short*)Cout)[(size_t)m * N + n] = f2bf(v);
            else
                ((float*)Cout)[(size_t)m * N + n] = v;
        }
    }
}

// attention: per-batch block. scores over L=196, softmax, context -> xc ctx slot (bf16)
__global__ __launch_bounds__(256)
void k_attention(const unsigned short* __restrict__ enc_bf,  // [B,L,D] bf16
                 const float* __restrict__ q,                // [B,D]
                 unsigned short* __restrict__ xc) {
    __shared__ float qs[D_];
    __shared__ float sc[L_];
    __shared__ float red[256];
    const int b = blockIdx.x, tid = threadIdx.x;

    for (int d = tid; d < D_; d += 256) qs[d] = q[(long)b * D_ + d];
    __syncthreads();

    for (int l = tid; l < L_; l += 256) {
        const unsigned short* row = enc_bf + ((size_t)b * L_ + l) * D_;
        float acc = 0.0f;
        for (int d = 0; d < D_; d += 8) {
            uint4 p = *(const uint4*)(row + d);
            const unsigned short* ph = (const unsigned short*)&p;
#pragma unroll
            for (int i = 0; i < 8; ++i) acc += bf2f(ph[i]) * qs[d + i];
        }
        sc[l] = acc;
    }
    __syncthreads();

    red[tid] = (tid < L_) ? sc[tid] : -3.4e38f;
    __syncthreads();
    for (int s = 128; s > 0; s >>= 1) {
        if (tid < s) red[tid] = fmaxf(red[tid], red[tid + s]);
        __syncthreads();
    }
    float mx = red[0];
    __syncthreads();
    red[tid] = (tid < L_) ? __expf(sc[tid] - mx) : 0.0f;
    __syncthreads();
    for (int s = 128; s > 0; s >>= 1) {
        if (tid < s) red[tid] += red[tid + s];
        __syncthreads();
    }
    float inv = 1.0f / red[0];
    __syncthreads();
    if (tid < L_) sc[tid] = __expf(sc[tid] - mx) * inv;
    __syncthreads();

    for (int d = tid; d < D_; d += 256) {
        float acc = 0.0f;
        const unsigned short* col = enc_bf + (size_t)b * L_ * D_ + d;
        for (int l = 0; l < L_; ++l) acc += sc[l] * bf2f(col[(size_t)l * D_]);
        xc[(size_t)b * KXC + E_ + d] = f2bf(acc);
    }
}

// LSTM cell pointwise (torch gate order i,f,g,o); writes h_new bf16 into xc h-slot
__global__ void k_lstm_cell(const float* __restrict__ gates,  // [B,4D]
                            float* __restrict__ c,            // [B,D] in/out
                            unsigned short* __restrict__ xc) {
    int i = blockIdx.x * blockDim.x + threadIdx.x;  // 64*1024
    if (i >= B_ * D_) return;
    int b = i >> 10, d = i & (D_ - 1);
    const float* g = gates + (long)b * 4 * D_;
    float ig = 1.0f / (1.0f + __expf(-g[d]));
    float fg = 1.0f / (1.0f + __expf(-g[D_ + d]));
    float gg = tanhf(g[2 * D_ + d]);
    float og = 1.0f / (1.0f + __expf(-g[3 * D_ + d]));
    float cn = fg * c[i] + ig * gg;
    c[i] = cn;
    float hn = og * tanhf(cn);
    xc[(long)b * KXC + KIH + d] = f2bf(hn);
}

// log-softmax over V=10000 per batch row -> d_out[b, t, :]
__global__ __launch_bounds__(256)
void k_log_softmax(const float* __restrict__ lp,  // [B,V]
                   float* __restrict__ out, int t) {
    __shared__ float red[256];
    const int b = blockIdx.x, tid = threadIdx.x;
    const float* row = lp + (size_t)b * V_;
    float mx = -3.4e38f;
    for (int v = tid; v < V_; v += 256) mx = fmaxf(mx, row[v]);
    red[tid] = mx; __syncthreads();
    for (int s = 128; s > 0; s >>= 1) {
        if (tid < s) red[tid] = fmaxf(red[tid], red[tid + s]);
        __syncthreads();
    }
    mx = red[0]; __syncthreads();
    float sm = 0.0f;
    for (int v = tid; v < V_; v += 256) sm += __expf(row[v] - mx);
    red[tid] = sm; __syncthreads();
    for (int s = 128; s > 0; s >>= 1) {
        if (tid < s) red[tid] += red[tid + s];
        __syncthreads();
    }
    float lse = mx + logf(red[0]);
    float* o = out + ((size_t)b * T_ + t) * V_;
    for (int v = tid; v < V_; v += 256) o[v] = row[v] - lse;
}

// ------------------------------------------------------------------ host side
static inline size_t align256(size_t x) { return (x + 255) & ~(size_t)255; }

extern "C" void kernel_launch(void* const* d_in, const int* in_sizes, int n_in,
                              void* d_out, int out_size, void* d_ws, size_t ws_size,
                              hipStream_t stream) {
    (void)in_sizes; (void)n_in; (void)out_size; (void)ws_size;
    const float* enc    = (const float*)d_in[0];
    const int*   tokens = (const int*)  d_in[1];
    const float* h0     = (const float*)d_in[2];
    const float* c0     = (const float*)d_in[3];
    const float* emb    = (const float*)d_in[4];
    const float* W_attn = (const float*)d_in[5];
    const float* b_attn = (const float*)d_in[6];
    const float* W_ih   = (const float*)d_in[7];
    const float* b_ih   = (const float*)d_in[8];
    const float* W_hh   = (const float*)d_in[9];
    const float* b_hh   = (const float*)d_in[10];
    const float* W_proj = (const float*)d_in[11];
    const float* b_proj = (const float*)d_in[12];
    const float* W_logit= (const float*)d_in[13];
    const float* b_logit= (const float*)d_in[14];
    float* out = (float*)d_out;

    char* ws = (char*)d_ws;
    size_t off = 0;
    auto take = [&](size_t bytes) { char* p = ws + off; off += align256(bytes); return p; };
    unsigned short* Wg_bf     = (unsigned short*)take((size_t)4 * D_ * KXC * 2);   // 20 MB
    unsigned short* Wattn_bf  = (unsigned short*)take((size_t)D_ * D_ * 2);        //  2 MB
    unsigned short* Wproj_bf  = (unsigned short*)take((size_t)D_ * KXC * 2);       //  5 MB
    unsigned short* Wlogit_bf = (unsigned short*)take((size_t)V_ * D_ * 2);        // 20 MB
    unsigned short* emb_bf    = (unsigned short*)take((size_t)V_ * E_ * 2);        // 10 MB
    unsigned short* enc_bf    = (unsigned short*)take((size_t)B_ * L_ * D_ * 2);   // 26 MB
    float*          bias_g    = (float*)take((size_t)4 * D_ * 4);
    unsigned short* xc        = (unsigned short*)take((size_t)B_ * KXC * 2);
    float*          qbuf      = (float*)take((size_t)B_ * D_ * 4);
    float*          gates     = (float*)take((size_t)B_ * 4 * D_ * 4);
    float*          c_ws      = (float*)take((size_t)B_ * D_ * 4);
    unsigned short* proj_bf   = (unsigned short*)take((size_t)B_ * D_ * 2);
    float*          logit_pre = (float*)take((size_t)B_ * V_ * 4);

    auto blocks = [](long n) { return (unsigned)((n + 255) / 256); };

    // ---- prep: bf16 conversions (deterministic, re-done each launch)
    k_f32_to_bf16<<<blocks((long)D_ * D_),   256, 0, stream>>>(Wattn_bf,  W_attn, (long)D_ * D_);
    k_f32_to_bf16<<<blocks((long)D_ * KXC),  256, 0, stream>>>(Wproj_bf,  W_proj, (long)D_ * KXC);
    k_f32_to_bf16<<<blocks((long)V_ * D_),   256, 0, stream>>>(Wlogit_bf, W_logit,(long)V_ * D_);
    k_f32_to_bf16<<<blocks((long)V_ * E_),   256, 0, stream>>>(emb_bf,    emb,    (long)V_ * E_);
    k_f32_to_bf16<<<blocks((long)B_*L_*D_),  256, 0, stream>>>(enc_bf,    enc,    (long)B_ * L_ * D_);
    k_build_wg   <<<blocks((long)4*D_*KXC),  256, 0, stream>>>(Wg_bf, W_ih, W_hh, (long)4 * D_ * KXC);
    k_bias_sum   <<<blocks(4 * D_),          256, 0, stream>>>(bias_g, b_ih, b_hh, 4 * D_);
    k_init_state <<<blocks(B_ * D_),         256, 0, stream>>>(c_ws, xc, h0, c0);

    // ---- recurrent steps
    for (int t = 0; t < T_; ++t) {
        k_embed<<<blocks(B_ * E_), 256, 0, stream>>>(xc, emb_bf, tokens, t);
        // q = h @ W_attn^T + b_attn     (A = xc h-slot, lda = 2560)
        k_gemm_wmma<false><<<D_ / 64, 128, 0, stream>>>(xc + KIH, KXC, Wattn_bf, b_attn,
                                                        qbuf, D_, D_);
        k_attention<<<B_, 256, 0, stream>>>(enc_bf, qbuf, xc);
        // gates = [e,ctx,h] @ [W_ih|W_hh]^T + (b_ih+b_hh)
        k_gemm_wmma<false><<<(4 * D_) / 64, 128, 0, stream>>>(xc, KXC, Wg_bf, bias_g,
                                                              gates, 4 * D_, KXC);
        k_lstm_cell<<<blocks(B_ * D_), 256, 0, stream>>>(gates, c_ws, xc);
        // proj = [e,ctx,h_new] @ W_proj^T + b_proj   (stored bf16 for next GEMM)
        k_gemm_wmma<true><<<D_ / 64, 128, 0, stream>>>(xc, KXC, Wproj_bf, b_proj,
                                                       proj_bf, D_, KXC);
        // logits_pre = proj @ W_logit^T + b_logit   (N=10000 -> 157 blocks, tail masked)
        k_gemm_wmma<false><<<(V_ + 63) / 64, 128, 0, stream>>>(proj_bf, D_, Wlogit_bf, b_logit,
                                                               logit_pre, V_, D_);
        k_log_softmax<<<B_, 256, 0, stream>>>(logit_pre, out, t);
    }
}
